// DiffusionMSAE_91001767068443
// MI455X (gfx1250) — matrix operand order repaired and checked by reading, not compiled, not run on hardware
//
#include <hip/hip_runtime.h>
#include <hip/hip_bf16.h>
#include <stdint.h>

typedef __bf16        v16bf __attribute__((ext_vector_type(16)));
typedef float         v8f   __attribute__((ext_vector_type(8)));
typedef unsigned int  v8u   __attribute__((ext_vector_type(8)));

#define IN_C  1280
#define HID   5120
#define NTOK  8192      // 32 * 16 * 16
#define BATCH 32
#define PIX   256       // 16*16
#define KSTEPS (IN_C / 32)   // 40

__device__ __forceinline__ unsigned short f2bf(float f) {
    unsigned int u = __float_as_uint(f);
    u += 0x7FFFu + ((u >> 16) & 1u);   // round-to-nearest-even
    return (unsigned short)(u >> 16);
}

__device__ __forceinline__ v16bf make_frag(uint4 lo, uint4 hi) {
    v8u t;
    t[0] = lo.x; t[1] = lo.y; t[2] = lo.z; t[3] = lo.w;
    t[4] = hi.x; t[5] = hi.y; t[6] = hi.z; t[7] = hi.w;
    return __builtin_bit_cast(v16bf, t);
}

// x[B][C][H][W] f32 -> xbf[n][c] bf16, n = b*256 + (h*16+w)
__global__ void pack_x_kernel(const float* __restrict__ x, unsigned short* __restrict__ xbf) {
    size_t i = (size_t)blockIdx.x * blockDim.x + threadIdx.x;
    if (i >= (size_t)BATCH * IN_C * PIX) return;
    size_t p = i % PIX;
    size_t c = (i / PIX) % IN_C;
    size_t b = i / ((size_t)PIX * IN_C);
    xbf[(b * PIX + p) * IN_C + c] = f2bf(x[i]);
}

// W_enc[k][j] f32 -> WT[j][k] bf16 (columns of W_enc become contiguous rows)
__global__ void pack_w_kernel(const float* __restrict__ W, unsigned short* __restrict__ WT) {
    size_t i = (size_t)blockIdx.x * blockDim.x + threadIdx.x;
    if (i >= (size_t)IN_C * HID) return;
    size_t j = i % HID;
    size_t k = i / HID;
    WT[j * IN_C + k] = f2bf(W[i]);
}

// encoded[8192][5120] = relu(xbf @ W + b_enc), bf16 WMMA, f32 accumulate.
// 128x128 block tile, double-buffered LDS, one barrier per K-step.
__global__ __launch_bounds__(256) void enc_gemm_kernel(
    const unsigned short* __restrict__ xbf,   // [NTOK][IN_C] bf16
    const unsigned short* __restrict__ WT,    // [HID][IN_C]  bf16
    const float* __restrict__ b_enc,
    float* __restrict__ enc)                  // [NTOK][HID] f32
{
    __shared__ alignas(16) unsigned short As[2][128][32];   // 2 x 8 KB
    __shared__ alignas(16) unsigned short Bs[2][128][32];   // 2 x 8 KB

    const int tid   = threadIdx.x;
    const int m0    = blockIdx.x * 128;
    const int n0    = blockIdx.y * 128;
    const int lrow  = tid >> 1;          // 0..127  (tile row to load)
    const int lhalf = tid & 1;           // K half: 0 -> K0..15, 1 -> K16..31

    const int lane   = tid & 31;
    const int wv     = tid >> 5;         // wave 0..7 -> 16-row M strip
    const int lm     = lane & 15;
    const int lchunk = lane >> 4;        // 0/1

    const unsigned short* gA = xbf + (size_t)(m0 + lrow) * IN_C + lhalf * 16;
    const unsigned short* gB = WT  + (size_t)(n0 + lrow) * IN_C + lhalf * 16;

    v8f acc[8];
    const v8f vzero = {0.f, 0.f, 0.f, 0.f, 0.f, 0.f, 0.f, 0.f};
#pragma unroll
    for (int t = 0; t < 8; ++t) acc[t] = vzero;

    // prologue: stage K-step 0 into buffer 0
    {
        const uint4* pa = (const uint4*)gA;
        const uint4* pb = (const uint4*)gB;
        uint4 a0 = pa[0], a1 = pa[1], b0 = pb[0], b1 = pb[1];
        uint4* sa = (uint4*)&As[0][lrow][lhalf * 16];
        sa[0] = a0; sa[1] = a1;
        uint4* sb = (uint4*)&Bs[0][lrow][lhalf * 16];
        sb[0] = b0; sb[1] = b1;
    }

    const int am = wv * 16 + lm;
    uint4 ra0, ra1, rb0, rb1;

    for (int it = 0; it < KSTEPS; ++it) {
        const int buf = it & 1;

        // prefetch next tile into registers (overlaps with compute below)
        if (it < KSTEPS - 1) {
            const uint4* pa = (const uint4*)(gA + (it + 1) * 32);
            ra0 = pa[0]; ra1 = pa[1];
            const uint4* pb = (const uint4*)(gB + (it + 1) * 32);
            rb0 = pb[0]; rb1 = pb[1];
        }

        __syncthreads();   // buf stores visible; prior reads of buf^1 done

        // A fragment: lane l<16 -> K {0..7,16..23}; lane>=16 -> K {8..15,24..31}
        uint4 alo = *(const uint4*)&As[buf][am][lchunk * 8];
        uint4 ahi = *(const uint4*)&As[buf][am][16 + lchunk * 8];
        v16bf afrag = make_frag(alo, ahi);

        // all 8 B fragments first, then back-to-back WMMAs
        v16bf bfr[8];
#pragma unroll
        for (int nt = 0; nt < 8; ++nt) {
            uint4 blo = *(const uint4*)&Bs[buf][nt * 16 + lm][lchunk * 8];
            uint4 bhi = *(const uint4*)&Bs[buf][nt * 16 + lm][16 + lchunk * 8];
            bfr[nt] = make_frag(blo, bhi);
        }
#pragma unroll
        for (int nt = 0; nt < 8; ++nt) {
            acc[nt] = __builtin_amdgcn_wmma_f32_16x16x32_bf16(
                false, afrag, false, bfr[nt], (short)0, acc[nt], false, false);
        }

        // stage prefetched tile into the other buffer
        if (it < KSTEPS - 1) {
            uint4* sa = (uint4*)&As[buf ^ 1][lrow][lhalf * 16];
            sa[0] = ra0; sa[1] = ra1;
            uint4* sb = (uint4*)&Bs[buf ^ 1][lrow][lhalf * 16];
            sb[0] = rb0; sb[1] = rb1;
        }
    }

    // epilogue: bias + relu, dense f32 store
    const int rowBase = m0 + wv * 16 + lchunk * 8;
#pragma unroll
    for (int nt = 0; nt < 8; ++nt) {
        const int col = n0 + nt * 16 + lm;
        const float bias = b_enc[col];
#pragma unroll
        for (int v = 0; v < 8; ++v) {
            float val = fmaxf(acc[nt][v] + bias, 0.0f);
            enc[(size_t)(rowBase + v) * HID + col] = val;
        }
    }
}

// per-token: top-32 selection, write sparse feats (k=16 and k=32), sparse decode
__global__ __launch_bounds__(256) void topk_decode_kernel(
    const float* __restrict__ Wdec,   // [HID][IN_C]
    const float* __restrict__ bdec,   // [IN_C]
    float* __restrict__ out)
{
    const int n   = blockIdx.x;
    const int tid = threadIdx.x;
    float* feats0 = out;
    float* feats1 = out + (size_t)NTOK * HID;   // also holds dense `encoded` as scratch
    float* recons = out + (size_t)2 * NTOK * HID;

    // row into registers (values are >= 0 after relu)
    float v[20];
    const float* encRow = feats1 + (size_t)n * HID;
#pragma unroll
    for (int j = 0; j < 20; ++j) v[j] = encRow[tid + 256 * j];

    __shared__ float redV[256];
    __shared__ int   redI[256];
    __shared__ float topV[32];
    __shared__ int   topI[32];

    for (int it = 0; it < 32; ++it) {
        float bm = -1.0f; int bj = -1;
#pragma unroll
        for (int j = 0; j < 20; ++j)
            if (v[j] > bm) { bm = v[j]; bj = j; }
        redV[tid] = bm;
        redI[tid] = (bj >= 0) ? (tid + 256 * bj) : tid;
        __syncthreads();
        for (int s = 128; s > 0; s >>= 1) {
            if (tid < s && redV[tid + s] > redV[tid]) {
                redV[tid] = redV[tid + s]; redI[tid] = redI[tid + s];
            }
            __syncthreads();
        }
        if (tid == 0) { topV[it] = fmaxf(redV[0], 0.0f); topI[it] = redI[0]; }
        __syncthreads();
        int wi = topI[it];
        if ((wi & 255) == tid) v[wi >> 8] = -1.0f;   // remove winner
        __syncthreads();
    }

    // feats: zero-fill both planes, then scatter top-k values
    float* f0 = feats0 + (size_t)n * HID;
    float* f1 = feats1 + (size_t)n * HID;
#pragma unroll
    for (int j = 0; j < 20; ++j) { f0[tid + 256 * j] = 0.0f; f1[tid + 256 * j] = 0.0f; }
    __syncthreads();
    if (tid < 16) f0[topI[tid]] = topV[tid];
    if (tid < 32) f1[topI[tid]] = topV[tid];

    // sparse decode: 5 channels per thread
    float acc[5];
#pragma unroll
    for (int u = 0; u < 5; ++u) acc[u] = bdec[tid + 256 * u];

#pragma unroll 1
    for (int i = 0; i < 16; ++i) {
        float val = topV[i];
        const float* wr = Wdec + (size_t)topI[i] * IN_C;
#pragma unroll
        for (int u = 0; u < 5; ++u) acc[u] = fmaf(val, wr[tid + 256 * u], acc[u]);
    }
    const size_t b = (size_t)(n >> 8), p = (size_t)(n & 255);
    float* r0 = recons + b * ((size_t)IN_C * PIX) + p;
#pragma unroll
    for (int u = 0; u < 5; ++u) r0[(size_t)(tid + 256 * u) * PIX] = acc[u];

#pragma unroll 1
    for (int i = 16; i < 32; ++i) {
        float val = topV[i];
        const float* wr = Wdec + (size_t)topI[i] * IN_C;
#pragma unroll
        for (int u = 0; u < 5; ++u) acc[u] = fmaf(val, wr[tid + 256 * u], acc[u]);
    }
    float* r1 = r0 + (size_t)BATCH * IN_C * PIX;
#pragma unroll
    for (int u = 0; u < 5; ++u) r1[(size_t)(tid + 256 * u) * PIX] = acc[u];
}

extern "C" void kernel_launch(void* const* d_in, const int* in_sizes, int n_in,
                              void* d_out, int out_size, void* d_ws, size_t ws_size,
                              hipStream_t stream) {
    const float* x     = (const float*)d_in[0];
    const float* W_enc = (const float*)d_in[1];
    const float* b_enc = (const float*)d_in[2];
    const float* W_dec = (const float*)d_in[3];
    const float* b_dec = (const float*)d_in[4];
    float* out = (float*)d_out;

    unsigned short* xbf = (unsigned short*)d_ws;                 // 8192*1280 bf16 = 20.97 MB
    unsigned short* WT  = xbf + (size_t)NTOK * IN_C;             // 5120*1280 bf16 = 13.1 MB

    pack_x_kernel<<<(BATCH * IN_C * PIX + 255) / 256, 256, 0, stream>>>(x, xbf);
    pack_w_kernel<<<(IN_C * HID + 255) / 256, 256, 0, stream>>>(W_enc, WT);

    float* enc = out + (size_t)NTOK * HID;   // dense encoded staged in feats[1] plane
    enc_gemm_kernel<<<dim3(NTOK / 128, HID / 128), 256, 0, stream>>>(xbf, WT, b_enc, enc);

    topk_decode_kernel<<<NTOK, 256, 0, stream>>>(W_dec, b_dec, out);
}
